// IAFVAEModel_84997402788012
// MI455X (gfx1250) — matrix operand order repaired and checked by reading, not compiled, not run on hardware
//
#include <hip/hip_runtime.h>
#include <hip/hip_bf16.h>
#include <math.h>

// MI455X / gfx1250 IAF-VAE forward pass.
// - All convs (incl. stride-1 "deconvs" == SAME convs) and dense layers run as
//   implicit GEMM on v_wmma_f32_16x16x32_f16 with f32 accumulation.
// - Conv kernels templated on (CIN, COUT, H, W): K-loop fully unrolls, A/B
//   fragments are contiguous b128 loads per the CDNA5 WMMA VGPR layouts.
// - Weight slabs are staged to LDS via gfx1250 async global->LDS DMA
//   (global_load_async_to_lds_b128 / s_wait_asynccnt) and reused by all 8
//   waves of the block from ds_load_b128.

typedef _Float16 f16;
typedef __attribute__((ext_vector_type(16))) _Float16 v16h;
typedef __attribute__((ext_vector_type(8)))  _Float16 v8h;
typedef __attribute__((ext_vector_type(8)))  float    v8f;

__device__ __forceinline__ float elu_f(float x) { return x > 0.f ? x : (expf(x) - 1.f); }

// ---------------- elementwise helpers ----------------
__global__ void k_f32_to_f16(const float* __restrict__ in, f16* __restrict__ out, int n) {
  int i = blockIdx.x * blockDim.x + threadIdx.x;
  if (i < n) out[i] = (f16)in[i];
}

// W f32 [T][Cin][Cout] -> Wt f16 [T][Cout][Cin]  (contiguous-K for WMMA B loads)
__global__ void k_transpose_w(const float* __restrict__ W, f16* __restrict__ Wt,
                              int T, int Cin, int Cout) {
  int i = blockIdx.x * blockDim.x + threadIdx.x;
  int tot = T * Cin * Cout;
  if (i >= tot) return;
  int t = i / (Cin * Cout);
  int r = i - t * Cin * Cout;
  int ci = r / Cout;
  int co = r - ci * Cout;
  Wt[(t * Cout + co) * Cin + ci] = (f16)W[i];
}

// 2x2 max pool, NHWC f16
__global__ void k_pool2x2(const f16* __restrict__ in, f16* __restrict__ out,
                          int N, int H, int W, int C) {
  int i = blockIdx.x * blockDim.x + threadIdx.x;
  int Ho = H >> 1, Wo = W >> 1;
  int tot = N * Ho * Wo * C;
  if (i >= tot) return;
  int c = i % C;
  int t = i / C;
  int wo = t % Wo; t /= Wo;
  int ho = t % Ho;
  int n = t / Ho;
  const f16* base = in + (((size_t)n * H + 2 * ho) * W + 2 * wo) * C + c;
  float a = (float)base[0];
  float b = (float)base[C];
  float d0 = (float)base[(size_t)W * C];
  float d1 = (float)base[(size_t)W * C + C];
  out[i] = (f16)fmaxf(fmaxf(a, b), fmaxf(d0, d1));
}

__global__ void k_bias_elu_f16(const float* __restrict__ acc, const float* __restrict__ bias,
                               f16* __restrict__ out, int M, int N) {
  int i = blockIdx.x * blockDim.x + threadIdx.x;
  if (i >= M * N) return;
  int n = i % N;
  out[i] = (f16)elu_f(acc[i] + bias[n]);
}

// -------- fragment pack helpers (documented CDNA5 WMMA VGPR layouts) --------
__device__ __forceinline__ v16h pack16(v8h lo, v8h hi8) {
  v16h r;
#pragma unroll
  for (int p = 0; p < 8; ++p) { r[p] = lo[p]; r[p + 8] = hi8[p]; }
  return r;
}

// ---------------- 3x3 SAME conv, LDS-staged weights (CIN>=32, COUT>=16) ----
// act: NHWC f16; wt: [tap(9)][COUT][CIN] f16.
// Block = 8 waves: one N-tile shared by block, 8 consecutive M-tiles.
// Weight slab for the N-tile (9 x 16 x CIN halves) staged via async DMA.
template <int CIN, int COUT, int HH, int WW>
__global__ void __launch_bounds__(256)
k_conv3x3_wmma_lds(const f16* __restrict__ act, const f16* __restrict__ wt,
                   const float* __restrict__ bias,
                   f16* __restrict__ out_h, int NB) {
  static_assert(CIN >= 32 && (CIN & 31) == 0 && COUT >= 16 && (COUT & 15) == 0);
  constexpr int NTN = COUT / 16;
  constexpr int WHALVES = 9 * 16 * CIN;          // weight slab halves
  __shared__ f16 smem[WHALVES];

  const int tid = threadIdx.x;
  const int lane = tid & 31;
  const int wib = tid >> 5;                       // wave in block, 0..7
  const int nti = blockIdx.x % NTN;
  const int mti = (blockIdx.x / NTN) * 8 + wib;

  // ---- async-stage the weight slab: 16B per lane per instruction ----
  {
    const int c0 = nti * 16;
    const unsigned base = (unsigned)(uintptr_t)(&smem[0]);
    constexpr int CHK = WHALVES / 8;              // 16-byte chunks
    for (int q = tid; q < CHK; q += 256) {
      int tap = q / (2 * CIN);
      int r = q - tap * (2 * CIN);
      const f16* src = wt + ((size_t)tap * COUT + c0) * CIN + (size_t)r * 8;
      unsigned dst = base + (unsigned)q * 16;
      asm volatile("global_load_async_to_lds_b128 %0, %1, off"
                   :: "v"(dst), "v"(src) : "memory");
    }
    asm volatile("s_wait_asynccnt 0x0" ::: "memory");
    __syncthreads();
  }

  const int lh = lane & 15;
  const bool hi = lane >= 16;
  const int m = mti * 16 + lh;                    // flattened (n,h,w); WW%16==0
  const int wx0 = m & (WW - 1);
  const int hy0 = (m / WW) & (HH - 1);
  const int n = m / (WW * HH);

  v8f acc = {};
  constexpr int CHUNKS = 9 * CIN / 32;
#pragma unroll
  for (int ch = 0; ch < CHUNKS; ++ch) {
    const int kb = ch * 32;                       // constants after unroll
    const int tap = kb / CIN;
    const int cb = kb - tap * CIN;
    const int kh = tap / 3, kw = tap - kh * 3;
    const int hy = hy0 + kh - 1, wx = wx0 + kw - 1;
    // A: lanes 0-15 hold K {0..7,16..23}, lanes 16-31 {8..15,24..31}
    v16h a = {};
    if (hy >= 0 && hy < HH && wx >= 0 && wx < WW) {
      const f16* pA = act + (((size_t)n * HH + hy) * WW + wx) * CIN + cb + (hi ? 8 : 0);
      a = pack16(*(const v8h*)pA, *(const v8h*)(pA + 16));
    }
    // B from LDS: lane = column; lanes 0-15 K {0..7,16..23}, lanes 16-31 {8..15,24..31}
    const f16* pB = &smem[(tap * 16 + lh) * CIN + cb + (hi ? 8 : 0)];
    v16h b = pack16(*(const v8h*)pB, *(const v8h*)(pB + 16));
    acc = __builtin_amdgcn_wmma_f32_16x16x32_f16(false, a, false, b, (short)0, acc,
                                                 false, false);
  }

  const int col = nti * 16 + lh;
  const float bsv = bias[col];
#pragma unroll
  for (int v = 0; v < 8; ++v) {
    int row = mti * 16 + v + (hi ? 8 : 0);        // C/D: VGPR v -> M = v (+8 hi)
    out_h[(size_t)row * COUT + col] = (f16)elu_f(acc[v] + bsv);
  }
}

// ---------------- 3x3 SAME conv, direct-global path (CIN==1 or COUT<16) ----
template <int CIN, int HH, int WW>
__global__ void __launch_bounds__(256)
k_conv3x3_wmma(const f16* __restrict__ act, const f16* __restrict__ wt,
               const float* __restrict__ bias,
               f16* __restrict__ out_h, float* __restrict__ out_f,
               int NB, int Cout) {
  const int lane = threadIdx.x & 31;
  const int wave = (int)((blockIdx.x * blockDim.x + threadIdx.x) >> 5);
  const int M = NB * HH * WW;
  const int ntn = (Cout + 15) >> 4;
  if (wave >= (M >> 4) * ntn) return;            // wave-uniform: EXEC all-1s
  const int nti = wave % ntn, mti = wave / ntn;
  const int lh = lane & 15;
  const bool hi = lane >= 16;
  const int m = mti * 16 + lh;
  const int wx0 = m & (WW - 1);
  const int hy0 = (m / WW) & (HH - 1);
  const int n = m / (WW * HH);
  const int col = nti * 16 + lh;

  v8f acc = {};
  if constexpr (CIN == 1) {
    v16h a = {}, b = {};
#pragma unroll
    for (int p = 0; p < 16; ++p) {
      int k = (p < 8 ? p : p + 8) + (hi ? 8 : 0);
      f16 av = (f16)0.f;
      if (k < 9) {
        int kh = k / 3, kw = k - kh * 3;
        int hy = hy0 + kh - 1, wx = wx0 + kw - 1;
        if (hy >= 0 && hy < HH && wx >= 0 && wx < WW)
          av = act[((size_t)n * HH + hy) * WW + wx];
      }
      a[p] = av;
      int kk = (p < 8 ? p : p + 8) + (hi ? 8 : 0);
      f16 bv = (f16)0.f;
      if (kk < 9 && col < Cout) bv = wt[kk * Cout + col];
      b[p] = bv;
    }
    acc = __builtin_amdgcn_wmma_f32_16x16x32_f16(false, a, false, b, (short)0, acc,
                                                 false, false);
  } else {
    constexpr int CHUNKS = 9 * CIN / 32;
#pragma unroll
    for (int ch = 0; ch < CHUNKS; ++ch) {
      const int kb = ch * 32;
      const int tap = kb / CIN;
      const int cb = kb - tap * CIN;
      const int kh = tap / 3, kw = tap - kh * 3;
      const int hy = hy0 + kh - 1, wx = wx0 + kw - 1;
      v16h a = {};
      if (hy >= 0 && hy < HH && wx >= 0 && wx < WW) {
        const f16* pA = act + (((size_t)n * HH + hy) * WW + wx) * CIN + cb + (hi ? 8 : 0);
        a = pack16(*(const v8h*)pA, *(const v8h*)(pA + 16));
      }
      v16h b = {};
      if (col < Cout) {
        const f16* pB = wt + ((size_t)tap * Cout + col) * CIN + cb + (hi ? 8 : 0);
        b = pack16(*(const v8h*)pB, *(const v8h*)(pB + 16));
      }
      acc = __builtin_amdgcn_wmma_f32_16x16x32_f16(false, a, false, b, (short)0, acc,
                                                   false, false);
    }
  }

  if (col < Cout) {
    float bsv = bias[col];
#pragma unroll
    for (int v = 0; v < 8; ++v) {
      int row = mti * 16 + v + (hi ? 8 : 0);
      float y = elu_f(acc[v] + bsv);
      if (out_h) out_h[(size_t)row * Cout + col] = (f16)y;
      if (out_f) out_f[(size_t)row * Cout + col] = y;
    }
  }
}

// ---------------- plain GEMM on WMMA: D = A[MxK] * Bt[NxK]^T ----------------
// Preconditions (all call sites): M%16==0, N%16==0, K%32==0.
// MODE 0: out f16 = elu(acc+bias); MODE 1: out f32 = acc+bias;
// MODE 2: split-K f32 atomicAdd into out (no bias/act).
template <int MODE>
__global__ void __launch_bounds__(256)
k_gemm_wmma(const f16* __restrict__ A, const f16* __restrict__ Bt,
            const float* __restrict__ bias, void* __restrict__ out,
            int M, int K, int N, int kslices) {
  const int lane = threadIdx.x & 31;
  const int wgl = (int)((blockIdx.x * blockDim.x + threadIdx.x) >> 5);
  const int ntn = N >> 4;
  const int ntm = M >> 4;
  const int tiles = ntm * ntn;
  if (wgl >= tiles * kslices) return;            // wave-uniform exit
  const int slice = wgl / tiles;
  const int tile = wgl - slice * tiles;
  const int nti = tile % ntn, mti = tile / ntn;
  const int lh = lane & 15;
  const bool hi = lane >= 16;
  const int m = mti * 16 + lh;
  const int col = nti * 16 + lh;

  const int chunks = K >> 5;
  const int cps = (chunks + kslices - 1) / kslices;
  const int k0 = slice * cps * 32;
  int k1 = (slice * cps + cps) * 32;
  if (k1 > K) k1 = K;

  const f16* pA0 = A + (size_t)m * K + (hi ? 8 : 0);
  const f16* pB0 = Bt + (size_t)col * K + (hi ? 8 : 0);

  v8f acc = {};
#pragma unroll 2
  for (int kb = k0; kb < k1; kb += 32) {
    v16h a = pack16(*(const v8h*)(pA0 + kb), *(const v8h*)(pA0 + kb + 16));
    v16h b = pack16(*(const v8h*)(pB0 + kb), *(const v8h*)(pB0 + kb + 16));
    acc = __builtin_amdgcn_wmma_f32_16x16x32_f16(false, a, false, b, (short)0, acc,
                                                 false, false);
  }

#pragma unroll
  for (int v = 0; v < 8; ++v) {
    int row = mti * 16 + v + (hi ? 8 : 0);
    if constexpr (MODE == 2) {
      atomicAdd((float*)out + (size_t)row * N + col, acc[v]);
    } else if constexpr (MODE == 1) {
      ((float*)out)[(size_t)row * N + col] = acc[v] + bias[col];
    } else {
      ((f16*)out)[(size_t)row * N + col] = (f16)elu_f(acc[v] + bias[col]);
    }
  }
}

// ---------------- sampling + 2 IAF flows (inherently sequential) ----------------
// One wave32 block; lane = batch row. Only columns 0,1 of each flow layer are used.
__global__ void k_flows(const float* __restrict__ mean, const float* __restrict__ logvar,
                        const float* __restrict__ eps, const float* __restrict__ fp,
                        f16* __restrict__ z16) {
  __shared__ float mh[64 * 32];
  __shared__ float sh[64 * 32];
  const int b = threadIdx.x;      // 0..31
  float z[32];
#pragma unroll
  for (int d = 0; d < 32; ++d)
    z[d] = mean[b * 32 + d] + expf(0.5f * logvar[b * 32 + d]) * eps[b * 32 + d];

  const float* p = fp;
  for (int f = 0; f < 2; ++f) {
    for (int i = 0; i < 64; ++i) {
      int units = (i & 1) ? 2 : 64;
      int rows = 32 + i;
      float a0 = p[rows * units + 0];   // bias[0]
      float a1 = p[rows * units + 1];   // bias[1]
#pragma unroll
      for (int j = 0; j < 32; ++j) {
        a0 += z[j] * p[j * units + 0];
        a1 += z[j] * p[j * units + 1];
      }
      for (int j = 0; j < i; ++j) {
        float mj = mh[j * 32 + b];
        a0 += mj * p[(32 + j) * units + 0];
        a1 += mj * p[(32 + j) * units + 1];
      }
      if (!(i & 1)) { a0 = elu_f(a0); a1 = elu_f(a1); }
      mh[i * 32 + b] = a0;
      sh[i * 32 + b] = (a1 > 20.f) ? a1 : log1pf(expf(a1));   // softplus
      p += (size_t)(rows + 1) * units;
    }
    for (int i = 0; i < 64; ++i) {
      float mi = mh[i * 32 + b], si = sh[i * 32 + b];
      float e = expf(-si);
#pragma unroll
      for (int d = 0; d < 32; ++d)
        z[d] = ((z[d] - mi) / si) * e + mi;
    }
  }
#pragma unroll
  for (int d = 0; d < 32; ++d) z16[b * 32 + d] = (f16)z[d];
}

// ---------------- launch ----------------
extern "C" void kernel_launch(void* const* d_in, const int* in_sizes, int n_in,
                              void* d_out, int out_size, void* d_ws, size_t ws_size,
                              hipStream_t stream) {
  (void)in_sizes; (void)n_in; (void)out_size; (void)ws_size;

  const float* x    = (const float*)d_in[0];
  const float* eps  = (const float*)d_in[1];
  const float* w1   = (const float*)d_in[2];  const float* b1   = (const float*)d_in[3];
  const float* w2   = (const float*)d_in[4];  const float* b2   = (const float*)d_in[5];
  const float* w3   = (const float*)d_in[6];  const float* b3   = (const float*)d_in[7];
  const float* wfc  = (const float*)d_in[8];  const float* bfc  = (const float*)d_in[9];
  const float* wme  = (const float*)d_in[10]; const float* bme  = (const float*)d_in[11];
  const float* wlv  = (const float*)d_in[12]; const float* blv  = (const float*)d_in[13];
  // d_in[14 .. 269]: flow params (2 flows x 64 layers x {W,b})
  const float* wdec = (const float*)d_in[270]; const float* bdec = (const float*)d_in[271];
  const float* wd1  = (const float*)d_in[272]; const float* bd1  = (const float*)d_in[273];
  const float* wd2  = (const float*)d_in[274]; const float* bd2  = (const float*)d_in[275];
  const float* wout = (const float*)d_in[276]; const float* bout = (const float*)d_in[277];

  char* ws = (char*)d_ws;
  size_t off = 0;
  auto alloc = [&](size_t bytes) -> char* {
    char* p = ws + off;
    off = (off + bytes + 255) & ~(size_t)255;
    return p;
  };

  f16*   x16   = (f16*)alloc((size_t)524288 * 2);
  f16*   c1    = (f16*)alloc((size_t)32 * 128 * 128 * 32 * 2);
  f16*   p1    = (f16*)alloc((size_t)32 * 64 * 64 * 32 * 2);
  f16*   c2    = (f16*)alloc((size_t)32 * 64 * 64 * 64 * 2);
  f16*   p2    = (f16*)alloc((size_t)32 * 32 * 32 * 64 * 2);
  f16*   c3    = (f16*)alloc((size_t)32 * 32 * 32 * 128 * 2);
  f16*   p3    = (f16*)alloc((size_t)32 * 16 * 16 * 128 * 2);
  f16*   wt1   = (f16*)alloc(288 * 2);
  f16*   wt2   = (f16*)alloc(18432 * 2);
  f16*   wt3   = (f16*)alloc(73728 * 2);
  f16*   wtfc  = (f16*)alloc((size_t)1048576 * 2);
  f16*   wtm   = (f16*)alloc(1024 * 2);
  f16*   wtlv  = (f16*)alloc(1024 * 2);
  f16*   wtdec = (f16*)alloc((size_t)1048576 * 2);
  f16*   wtd1  = (f16*)alloc(73728 * 2);
  f16*   wtd2  = (f16*)alloc(18432 * 2);
  f16*   wtout = (f16*)alloc(288 * 2);
  float* accfc = (float*)alloc(1024 * 4);
  f16*   h16   = (f16*)alloc(1024 * 2);
  f16*   z16   = (f16*)alloc(1024 * 2);
  float* fpack = (float*)alloc((size_t)270464 * 4);   // 2 * 135232 packed flow params
  f16*   dfc   = (f16*)alloc((size_t)1048576 * 2);
  f16*   d1    = (f16*)alloc((size_t)524288 * 2);
  f16*   d2    = (f16*)alloc((size_t)262144 * 2);

  auto G = [](long n, int t) { return (int)((n + t - 1) / t); };
  auto convGrid = [](int M, int Cout) {
    long waves = (long)(M >> 4) * ((Cout + 15) >> 4);
    return (int)((waves * 32 + 255) / 256);
  };
  // LDS-staged conv: blocks = (Mtiles/8) * (COUT/16)
  auto convGridLds = [](int M, int Cout) { return ((M >> 4) / 8) * (Cout >> 4); };

  // --- f16 conversions / weight transposes ---
  k_f32_to_f16<<<G(524288, 256), 256, 0, stream>>>(x, x16, 524288);
  k_transpose_w<<<G(288, 256), 256, 0, stream>>>(w1, wt1, 9, 1, 32);
  k_transpose_w<<<G(18432, 256), 256, 0, stream>>>(w2, wt2, 9, 32, 64);
  k_transpose_w<<<G(73728, 256), 256, 0, stream>>>(w3, wt3, 9, 64, 128);
  k_transpose_w<<<G(1048576, 256), 256, 0, stream>>>(wfc, wtfc, 1, 32768, 32);
  k_transpose_w<<<G(1024, 256), 256, 0, stream>>>(wme, wtm, 1, 32, 32);
  k_transpose_w<<<G(1024, 256), 256, 0, stream>>>(wlv, wtlv, 1, 32, 32);
  k_transpose_w<<<G(1048576, 256), 256, 0, stream>>>(wdec, wtdec, 1, 32, 32768);
  k_transpose_w<<<G(73728, 256), 256, 0, stream>>>(wd1, wtd1, 9, 128, 64);
  k_transpose_w<<<G(18432, 256), 256, 0, stream>>>(wd2, wtd2, 9, 64, 32);
  k_transpose_w<<<G(288, 256), 256, 0, stream>>>(wout, wtout, 9, 32, 1);

  // --- encoder: conv -> pool x3 ---
  k_conv3x3_wmma<1, 128, 128><<<convGrid(524288, 32), 256, 0, stream>>>(
      x16, wt1, b1, c1, nullptr, 32, 32);
  k_pool2x2<<<G(4194304, 256), 256, 0, stream>>>(c1, p1, 32, 128, 128, 32);
  k_conv3x3_wmma_lds<32, 64, 64, 64><<<convGridLds(131072, 64), 256, 0, stream>>>(
      p1, wt2, b2, c2, 32);
  k_pool2x2<<<G(2097152, 256), 256, 0, stream>>>(c2, p2, 32, 64, 64, 64);
  k_conv3x3_wmma_lds<64, 128, 32, 32><<<convGridLds(32768, 128), 256, 0, stream>>>(
      p2, wt3, b3, c3, 32);
  k_pool2x2<<<G(1048576, 256), 256, 0, stream>>>(c3, p3, 32, 32, 32, 128);

  // --- enc_fc: split-K (M=N=32, K=32768) ---
  hipMemsetAsync(accfc, 0, 1024 * 4, stream);
  {
    const int kslices = 32, tiles = 4;
    k_gemm_wmma<2><<<(tiles * kslices * 32 + 255) / 256, 256, 0, stream>>>(
        p3, wtfc, nullptr, accfc, 32, 32768, 32, kslices);
  }
  k_bias_elu_f16<<<G(1024, 256), 256, 0, stream>>>(accfc, bfc, h16, 32, 32);

  float* out_recon = (float*)d_out;
  float* out_mean  = out_recon + 8192;
  float* out_lv    = out_recon + 9216;
  k_gemm_wmma<1><<<1, 256, 0, stream>>>(h16, wtm, bme, out_mean, 32, 32, 32, 1);
  k_gemm_wmma<1><<<1, 256, 0, stream>>>(h16, wtlv, blv, out_lv, 32, 32, 32, 1);

  // --- pack flow params contiguously (capture-legal D2D copies) ---
  {
    size_t o = 0;
    for (int f = 0; f < 2; ++f)
      for (int i = 0; i < 64; ++i) {
        int units = (i & 1) ? 2 : 64;
        int rows = 32 + i;
        hipMemcpyAsync((char*)fpack + o * 4, d_in[14 + f * 128 + i * 2],
                       (size_t)rows * units * 4, hipMemcpyDeviceToDevice, stream);
        o += (size_t)rows * units;
        hipMemcpyAsync((char*)fpack + o * 4, d_in[14 + f * 128 + i * 2 + 1],
                       (size_t)units * 4, hipMemcpyDeviceToDevice, stream);
        o += (size_t)units;
      }
  }
  k_flows<<<1, 32, 0, stream>>>(out_mean, out_lv, eps, fpack, z16);

  // --- decoder ---
  // dec_fc: M=32, K=32, N=32768 -> 4096 tiles
  k_gemm_wmma<0><<<(4096 * 32 + 255) / 256, 256, 0, stream>>>(
      z16, wtdec, bdec, dfc, 32, 32, 32768, 1);
  k_conv3x3_wmma_lds<128, 64, 16, 16><<<convGridLds(8192, 64), 256, 0, stream>>>(
      dfc, wtd1, bd1, d1, 32);
  k_conv3x3_wmma_lds<64, 32, 16, 16><<<convGridLds(8192, 32), 256, 0, stream>>>(
      d1, wtd2, bd2, d2, 32);
  k_conv3x3_wmma<32, 16, 16><<<convGrid(8192, 1), 256, 0, stream>>>(
      d2, wtout, bout, nullptr, out_recon, 32, 1);
}
